// GraphAttentionEmbedding_48842368090474
// MI455X (gfx1250) — matrix operand order repaired and checked by reading, not compiled, not run on hardware
//
#include <hip/hip_runtime.h>

#define N_NODES 10000
#define N_EDGES 160000

typedef __attribute__((ext_vector_type(16))) _Float16 v16h;
typedef __attribute__((ext_vector_type(8)))  _Float16 v8h;
typedef __attribute__((ext_vector_type(8)))  float    v8f;
typedef __attribute__((ext_vector_type(4)))  int      v4i;

// ---- order-preserving float <-> uint encoding for atomicMax-based segment max ----
__device__ __forceinline__ unsigned gae_fenc(float f) {
  unsigned u = __float_as_uint(f);
  return (u & 0x80000000u) ? ~u : (u | 0x80000000u);
}
__device__ __forceinline__ float gae_fdec(unsigned u) {
  return __uint_as_float((u & 0x80000000u) ? (u & 0x7FFFFFFFu) : ~u);
}

// ---------------- prep kernels ----------------
__global__ void gae_f32_to_f16(const float* __restrict__ in, _Float16* __restrict__ out,
                               long long n) {
  long long i = (long long)blockIdx.x * blockDim.x + threadIdx.x;
  if (i < n) out[i] = (_Float16)in[i];
}

// W is [K, Nout] row-major f32; Wt is [Nout, K] row-major f16 (i.e. B column-major)
__global__ void gae_pack_wt(const float* __restrict__ W, _Float16* __restrict__ Wt,
                            int K, int Nout) {
  int i = blockIdx.x * blockDim.x + threadIdx.x;
  if (i >= K * Nout) return;
  int k = i / Nout, n = i % Nout;
  Wt[(size_t)n * K + k] = (_Float16)W[i];
}

// edge_attr[e, 0:128] = cos((last_update[src]-t[e]) * time_w + time_b); [128:256] = msg
__global__ void gae_edge_attr(const float* __restrict__ last_update,
                              const int* __restrict__ ei, const float* __restrict__ t,
                              const float* __restrict__ msg,
                              const float* __restrict__ time_w,
                              const float* __restrict__ time_b,
                              _Float16* __restrict__ ea) {
  long long i = (long long)blockIdx.x * blockDim.x + threadIdx.x;
  if (i >= (long long)N_EDGES * 256) return;
  int e = (int)(i >> 8), d = (int)(i & 255);
  float v;
  if (d < 128) {
    int s = ei[e];
    float rel = last_update[s] - t[e];
    v = __cosf(rel * time_w[d] + time_b[d]);
  } else {
    v = msg[(size_t)e * 128 + (d - 128)];
  }
  ea[i] = (_Float16)v;
}

// ---------------- WMMA GEMM v2 ----------------
// C[M,N] = A[M,K=256](f16,row) * Bt[N,K](f16, column-major B) + bias.
// Block = 256 threads = 8 waves; block computes a 128-row M panel x (NT*16)-col N panel.
// B panel (NT*16 cols x 256 K = NT*8KB) is staged once into LDS with
// global_load_async_to_lds_b128 (ASYNCcnt path), shared by all 8 waves; each wave
// register-blocks NT N-tiles so one A fragment feeds NT v_wmma ops.
// grid = (ceil(M/128), tilesN/NT).
template <typename OutT, int NT>
__global__ __launch_bounds__(256) void gae_wmma_gemm(const _Float16* __restrict__ A,
                                                     const _Float16* __restrict__ Bt,
                                                     const float* __restrict__ bias,
                                                     OutT* __restrict__ C,
                                                     int M, int Nn, int K /* must be 256 */) {
  __shared__ __align__(32) _Float16 sB[NT * 16 * 256];
  const int tid  = threadIdx.x;
  const int lane = tid & 31;
  const int wave = tid >> 5;
  const int chunks = NT * 16 * 256 / 8;  // number of 16B chunks in the panel
  const _Float16* gB = Bt + (size_t)blockIdx.y * (NT * 16) * 256;

  // ---- cooperative stage of the B panel into LDS ----
#if defined(__gfx1250__) && __has_builtin(__builtin_amdgcn_global_load_async_to_lds_b128)
  for (int c = tid; c < chunks; c += 256) {
    __builtin_amdgcn_global_load_async_to_lds_b128(
        (__attribute__((address_space(1))) v4i*)(gB + (size_t)c * 8),
        (__attribute__((address_space(3))) v4i*)(&sB[c * 8]),
        /*offset=*/0, /*cpol=*/0);
  }
  asm volatile("s_wait_asynccnt 0x0" ::: "memory");
#else
  for (int c = tid; c < chunks; c += 256)
    *(v8h*)(&sB[c * 8]) = *(const v8h*)(gB + (size_t)c * 8);
#endif
  __syncthreads();

  const long long mtile = (long long)blockIdx.x * 8 + wave;   // per-wave M tile
  const int r  = lane & 15;   // A row / B col / D col within a tile
  const int hi = lane >> 4;   // lane-half -> K sub-block

  if (mtile < (long long)(M >> 4)) {
    const _Float16* arow = A + (size_t)(mtile * 16 + r) * 256;
    v8f zero = {};
    v8f acc[NT];
#pragma unroll
    for (int j = 0; j < NT; ++j) acc[j] = zero;

    for (int kb = 0; kb < 256; kb += 32) {
      __builtin_prefetch(arow + kb + 64, 0, 3);
      // A fragment: lane half hi holds contiguous K = kb+16*hi+{0..15}
      const int ka = kb + hi * 16;
      v8h a0 = *(const v8h*)(arow + ka);
      v8h a1 = *(const v8h*)(arow + ka + 8);
      v16h a;
#pragma unroll
      for (int q = 0; q < 8; ++q) { a[q] = a0[q]; a[q + 8] = a1[q]; }
#pragma unroll
      for (int j = 0; j < NT; ++j) {
        // B fragment from LDS: lane half hi holds contiguous K = kb+16*hi+{0..15}
        v16h b = *(const v16h*)(&sB[(j * 16 + r) * 256 + kb + hi * 16]);
        acc[j] = __builtin_amdgcn_wmma_f32_16x16x32_f16(false, a, false, b,
                                                        (short)0, acc[j], false, false);
      }
    }

    const int m0 = (int)mtile * 16 + hi * 8;  // D: VGPR j -> row j (+8 for upper lane half)
#pragma unroll
    for (int j = 0; j < NT; ++j) {
      const int n = blockIdx.y * (NT * 16) + j * 16 + r;
      const float bv = bias ? bias[n] : 0.0f;
#pragma unroll
      for (int jj = 0; jj < 8; ++jj)
        C[(size_t)(m0 + jj) * Nn + n] = (OutT)(acc[j][jj] + bv);
    }
  }
}

// ---------------- conv1 attention (8 heads x 32) ----------------
// node1 layout per node: [q(256) | k(256) | v(256) | skip(256)]
__global__ void gae_attn1_alpha(const int* __restrict__ ei, const float* __restrict__ node1,
                                const _Float16* __restrict__ e1h,
                                float* __restrict__ alpha, unsigned* __restrict__ amax) {
  long long i = (long long)blockIdx.x * blockDim.x + threadIdx.x;
  if (i >= (long long)N_EDGES * 8) return;
  int e = (int)(i >> 3), h = (int)(i & 7);
  int s = ei[e], d = ei[N_EDGES + e];
  const float* q = node1 + (size_t)d * 1024 + h * 32;
  const float* k = node1 + (size_t)s * 1024 + 256 + h * 32;
  const _Float16* eh = e1h + (size_t)e * 256 + h * 32;
  float acc = 0.f;
#pragma unroll
  for (int c = 0; c < 32; ++c) acc += q[c] * (k[c] + (float)eh[c]);
  acc *= 0.17677669529663687f;  // 1/sqrt(32)
  alpha[i] = acc;
  atomicMax(&amax[(size_t)d * 8 + h], gae_fenc(acc));
}

__global__ void gae_attn1_ex(const int* __restrict__ ei, const unsigned* __restrict__ amax,
                             float* __restrict__ alpha, float* __restrict__ den) {
  long long i = (long long)blockIdx.x * blockDim.x + threadIdx.x;
  if (i >= (long long)N_EDGES * 8) return;
  int e = (int)(i >> 3), h = (int)(i & 7);
  int d = ei[N_EDGES + e];
  float ex = __expf(alpha[i] - gae_fdec(amax[(size_t)d * 8 + h]));
  alpha[i] = ex;
  atomicAdd(&den[(size_t)d * 8 + h], ex);
}

__global__ void gae_agg1(const int* __restrict__ ei, const float* __restrict__ node1,
                         const _Float16* __restrict__ e1h, const float* __restrict__ ex,
                         const float* __restrict__ den, float* __restrict__ agg) {
  long long i = (long long)blockIdx.x * blockDim.x + threadIdx.x;
  if (i >= (long long)N_EDGES * 256) return;
  int e = (int)(i >> 8), c = (int)(i & 255), h = c >> 5;
  int s = ei[e], d = ei[N_EDGES + e];
  float attn = ex[(size_t)e * 8 + h] / (den[(size_t)d * 8 + h] + 1e-16f);
  float ve = node1[(size_t)s * 1024 + 512 + c] + (float)e1h[(size_t)e * 256 + c];
  atomicAdd(&agg[(size_t)d * 256 + c], attn * ve);
}

__global__ void gae_fin1(const float* __restrict__ agg, const float* __restrict__ node1,
                         _Float16* __restrict__ hh) {
  long long i = (long long)blockIdx.x * blockDim.x + threadIdx.x;
  if (i >= (long long)N_NODES * 256) return;
  int n = (int)(i >> 8), c = (int)(i & 255);
  float v = agg[i] + node1[(size_t)n * 1024 + 768 + c];
  hh[i] = (_Float16)fmaxf(v, 0.f);
}

// ---------------- conv2 attention (1 head x 32) ----------------
// node2 layout per node: [q2(32) | k2(32) | v2(32) | skip2(32)]
__global__ void gae_attn2_alpha(const int* __restrict__ ei, const float* __restrict__ node2,
                                const float* __restrict__ e2,
                                float* __restrict__ alpha2, unsigned* __restrict__ amax2) {
  int e = blockIdx.x * blockDim.x + threadIdx.x;
  if (e >= N_EDGES) return;
  int s = ei[e], d = ei[N_EDGES + e];
  const float* q = node2 + (size_t)d * 128;
  const float* k = node2 + (size_t)s * 128 + 32;
  const float* ep = e2 + (size_t)e * 32;
  float acc = 0.f;
#pragma unroll
  for (int c = 0; c < 32; ++c) acc += q[c] * (k[c] + ep[c]);
  acc *= 0.17677669529663687f;
  alpha2[e] = acc;
  atomicMax(&amax2[d], gae_fenc(acc));
}

__global__ void gae_attn2_ex(const int* __restrict__ ei, const unsigned* __restrict__ amax2,
                             float* __restrict__ alpha2, float* __restrict__ den2) {
  int e = blockIdx.x * blockDim.x + threadIdx.x;
  if (e >= N_EDGES) return;
  int d = ei[N_EDGES + e];
  float ex = __expf(alpha2[e] - gae_fdec(amax2[d]));
  alpha2[e] = ex;
  atomicAdd(&den2[d], ex);
}

__global__ void gae_agg2(const int* __restrict__ ei, const float* __restrict__ node2,
                         const float* __restrict__ e2, const float* __restrict__ ex2,
                         const float* __restrict__ den2, float* __restrict__ agg2) {
  long long i = (long long)blockIdx.x * blockDim.x + threadIdx.x;
  if (i >= (long long)N_EDGES * 32) return;
  int e = (int)(i >> 5), c = (int)(i & 31);
  int s = ei[e], d = ei[N_EDGES + e];
  float attn = ex2[e] / (den2[d] + 1e-16f);
  float ve = node2[(size_t)s * 128 + 64 + c] + e2[(size_t)e * 32 + c];
  atomicAdd(&agg2[(size_t)d * 32 + c], attn * ve);
}

__global__ void gae_fin2(const float* __restrict__ agg2, const float* __restrict__ node2,
                         float* __restrict__ out) {
  long long i = (long long)blockIdx.x * blockDim.x + threadIdx.x;
  if (i >= (long long)N_NODES * 32) return;
  int n = (int)(i >> 5), c = (int)(i & 31);
  out[i] = fmaxf(agg2[i] + node2[(size_t)n * 128 + 96 + c], 0.f);
}

// ---------------- launcher ----------------
static inline unsigned gae_nblk(long long n, int t) { return (unsigned)((n + t - 1) / t); }

extern "C" void kernel_launch(void* const* d_in, const int* in_sizes, int n_in,
                              void* d_out, int out_size, void* d_ws, size_t ws_size,
                              hipStream_t stream) {
  const float* x           = (const float*)d_in[0];
  const float* last_update = (const float*)d_in[1];
  const int*   ei          = (const int*)d_in[2];
  const float* t           = (const float*)d_in[3];
  const float* msg         = (const float*)d_in[4];
  const float* time_w      = (const float*)d_in[5];
  const float* time_b      = (const float*)d_in[6];
  const float* Wq1 = (const float*)d_in[7];  const float* bq1 = (const float*)d_in[8];
  const float* Wk1 = (const float*)d_in[9];  const float* bk1 = (const float*)d_in[10];
  const float* Wv1 = (const float*)d_in[11]; const float* bv1 = (const float*)d_in[12];
  const float* We1 = (const float*)d_in[13];
  const float* Ws1 = (const float*)d_in[14]; const float* bs1 = (const float*)d_in[15];
  const float* Wq2 = (const float*)d_in[16]; const float* bq2 = (const float*)d_in[17];
  const float* Wk2 = (const float*)d_in[18]; const float* bk2 = (const float*)d_in[19];
  const float* Wv2 = (const float*)d_in[20]; const float* bv2 = (const float*)d_in[21];
  const float* We2 = (const float*)d_in[22];
  const float* Ws2 = (const float*)d_in[23]; const float* bs2 = (const float*)d_in[24];
  float* out = (float*)d_out;

  // ---- workspace carve-up (256B aligned) ----
  char* ws = (char*)d_ws;
  size_t off = 0;
  auto alloc = [&](size_t bytes) -> char* {
    char* p = ws + off;
    off = (off + bytes + 255) & ~(size_t)255;
    return p;
  };
  _Float16* xh    = (_Float16*)alloc((size_t)N_NODES * 256 * 2);
  _Float16* w1t   = (_Float16*)alloc((size_t)1024 * 256 * 2);
  _Float16* we1t  = (_Float16*)alloc((size_t)256 * 256 * 2);
  _Float16* we2t  = (_Float16*)alloc((size_t)32 * 256 * 2);
  _Float16* w2t   = (_Float16*)alloc((size_t)128 * 256 * 2);
  float*    b1    = (float*)alloc(1024 * 4);
  float*    b2    = (float*)alloc(128 * 4);
  _Float16* ea    = (_Float16*)alloc((size_t)N_EDGES * 256 * 2);
  float*    node1 = (float*)alloc((size_t)N_NODES * 1024 * 4);
  _Float16* e1h   = (_Float16*)alloc((size_t)N_EDGES * 256 * 2);
  float*    e2    = (float*)alloc((size_t)N_EDGES * 32 * 4);
  float*    alpha1= (float*)alloc((size_t)N_EDGES * 8 * 4);
  unsigned* amax1 = (unsigned*)alloc((size_t)N_NODES * 8 * 4);
  float*    den1  = (float*)alloc((size_t)N_NODES * 8 * 4);
  float*    agg1  = (float*)alloc((size_t)N_NODES * 256 * 4);
  _Float16* hh    = (_Float16*)alloc((size_t)N_NODES * 256 * 2);
  float*    node2 = (float*)alloc((size_t)N_NODES * 128 * 4);
  float*    alpha2= (float*)alloc((size_t)N_EDGES * 4);
  unsigned* amax2 = (unsigned*)alloc((size_t)N_NODES * 4);
  float*    den2  = (float*)alloc((size_t)N_NODES * 4);
  float*    agg2  = (float*)alloc((size_t)N_NODES * 32 * 4);

  // ---- zero the reduction buffers (every call; deterministic) ----
  (void)hipMemsetAsync(amax1, 0, (size_t)N_NODES * 8 * 4, stream);
  (void)hipMemsetAsync(den1,  0, (size_t)N_NODES * 8 * 4, stream);
  (void)hipMemsetAsync(agg1,  0, (size_t)N_NODES * 256 * 4, stream);
  (void)hipMemsetAsync(amax2, 0, (size_t)N_NODES * 4, stream);
  (void)hipMemsetAsync(den2,  0, (size_t)N_NODES * 4, stream);
  (void)hipMemsetAsync(agg2,  0, (size_t)N_NODES * 32 * 4, stream);

  // ---- bias concatenation ----
  (void)hipMemcpyAsync(b1 + 0,   bq1, 256 * 4, hipMemcpyDeviceToDevice, stream);
  (void)hipMemcpyAsync(b1 + 256, bk1, 256 * 4, hipMemcpyDeviceToDevice, stream);
  (void)hipMemcpyAsync(b1 + 512, bv1, 256 * 4, hipMemcpyDeviceToDevice, stream);
  (void)hipMemcpyAsync(b1 + 768, bs1, 256 * 4, hipMemcpyDeviceToDevice, stream);
  (void)hipMemcpyAsync(b2 + 0,  bq2, 32 * 4, hipMemcpyDeviceToDevice, stream);
  (void)hipMemcpyAsync(b2 + 32, bk2, 32 * 4, hipMemcpyDeviceToDevice, stream);
  (void)hipMemcpyAsync(b2 + 64, bv2, 32 * 4, hipMemcpyDeviceToDevice, stream);
  (void)hipMemcpyAsync(b2 + 96, bs2, 32 * 4, hipMemcpyDeviceToDevice, stream);

  // ---- pack / convert ----
  gae_f32_to_f16<<<gae_nblk((long long)N_NODES * 256, 256), 256, 0, stream>>>(
      x, xh, (long long)N_NODES * 256);
  gae_pack_wt<<<gae_nblk(256 * 256, 256), 256, 0, stream>>>(Wq1, w1t + 0 * 65536, 256, 256);
  gae_pack_wt<<<gae_nblk(256 * 256, 256), 256, 0, stream>>>(Wk1, w1t + 1 * 65536, 256, 256);
  gae_pack_wt<<<gae_nblk(256 * 256, 256), 256, 0, stream>>>(Wv1, w1t + 2 * 65536, 256, 256);
  gae_pack_wt<<<gae_nblk(256 * 256, 256), 256, 0, stream>>>(Ws1, w1t + 3 * 65536, 256, 256);
  gae_pack_wt<<<gae_nblk(256 * 256, 256), 256, 0, stream>>>(We1, we1t, 256, 256);
  gae_pack_wt<<<gae_nblk(256 * 32, 256), 256, 0, stream>>>(We2, we2t, 256, 32);
  gae_pack_wt<<<gae_nblk(256 * 32, 256), 256, 0, stream>>>(Wq2, w2t + 0 * 8192, 256, 32);
  gae_pack_wt<<<gae_nblk(256 * 32, 256), 256, 0, stream>>>(Wk2, w2t + 1 * 8192, 256, 32);
  gae_pack_wt<<<gae_nblk(256 * 32, 256), 256, 0, stream>>>(Wv2, w2t + 2 * 8192, 256, 32);
  gae_pack_wt<<<gae_nblk(256 * 32, 256), 256, 0, stream>>>(Ws2, w2t + 3 * 8192, 256, 32);

  // ---- edge features ----
  gae_edge_attr<<<gae_nblk((long long)N_EDGES * 256, 256), 256, 0, stream>>>(
      last_update, ei, t, msg, time_w, time_b, ea);

  // ---- WMMA GEMMs (grid = (ceil(Mtiles/8), tilesN/NT)) ----
  // node1 = xh @ [Wq1|Wk1|Wv1|Ws1] + biases : [10000,1024]
  gae_wmma_gemm<float, 4><<<dim3(79, 16), 256, 0, stream>>>(
      xh, w1t, b1, node1, N_NODES, 1024, 256);
  // e1h = ea @ We1 : [160000,256] (f16 out)
  gae_wmma_gemm<_Float16, 4><<<dim3(1250, 4), 256, 0, stream>>>(
      ea, we1t, nullptr, e1h, N_EDGES, 256, 256);
  // e2 = ea @ We2 : [160000,32]
  gae_wmma_gemm<float, 2><<<dim3(1250, 1), 256, 0, stream>>>(
      ea, we2t, nullptr, e2, N_EDGES, 32, 256);

  // ---- conv1 attention + aggregation ----
  gae_attn1_alpha<<<gae_nblk((long long)N_EDGES * 8, 256), 256, 0, stream>>>(
      ei, node1, e1h, alpha1, amax1);
  gae_attn1_ex<<<gae_nblk((long long)N_EDGES * 8, 256), 256, 0, stream>>>(
      ei, amax1, alpha1, den1);
  gae_agg1<<<gae_nblk((long long)N_EDGES * 256, 256), 256, 0, stream>>>(
      ei, node1, e1h, alpha1, den1, agg1);
  gae_fin1<<<gae_nblk((long long)N_NODES * 256, 256), 256, 0, stream>>>(agg1, node1, hh);

  // ---- conv2 node GEMM: node2 = hh @ [Wq2|Wk2|Wv2|Ws2] + biases : [10000,128] ----
  gae_wmma_gemm<float, 4><<<dim3(79, 2), 256, 0, stream>>>(
      hh, w2t, b2, node2, N_NODES, 128, 256);

  // ---- conv2 attention + aggregation ----
  gae_attn2_alpha<<<gae_nblk(N_EDGES, 256), 256, 0, stream>>>(ei, node2, e2, alpha2, amax2);
  gae_attn2_ex<<<gae_nblk(N_EDGES, 256), 256, 0, stream>>>(ei, amax2, alpha2, den2);
  gae_agg2<<<gae_nblk((long long)N_EDGES * 32, 256), 256, 0, stream>>>(
      ei, node2, e2, alpha2, den2, agg2);
  gae_fin2<<<gae_nblk((long long)N_NODES * 32, 256), 256, 0, stream>>>(agg2, node2, out);
}
